// NeRF_81398220193843
// MI455X (gfx1250) — compile-verified, matrix operand order checked by reading, not run on hardware
//
#include <hip/hip_runtime.h>
#include <hip/hip_bf16.h>

// CDNA5 wave32 WMMA types
typedef __attribute__((ext_vector_type(16))) _Float16 v16h;
typedef __attribute__((ext_vector_type(8)))  _Float16 v8h;
typedef __attribute__((ext_vector_type(8)))  float    v8f;

#define B_    2
#define NV    4
#define IMH   240
#define IMW   240
#define FH    120
#define FW    120
#define DFEAT 32
#define NPB   131072          // points per batch (Nr*Np)
#define NPTS  262144          // B_*NPB

// workspace layout: [0,384)   = w2c matrices (8 x 12 floats)
//                   [512,...) = swizzled f16 weights (~156KB total)
#define WS_WGT_OFF 512
#define OFF_L0 0              // base: K=96(pad from 70), N=128 -> 24 frags
#define OFF_L1 24576          // 128x128 -> 32 frags
#define OFF_L2 57344
#define OFF_L3 90112
#define OFF_L4 122880
#define OFF_L5 155648         // head: K=128, N=16(pad from 3) -> 4 frags

// ---------------------------------------------------------------------------
// A-fragment (16x32 f16, documented ISA layout):
//   lane L: M = L%16, g = L/16; half h: K = kc*32 + (h<8 ? h : h+8) + 8*g
// -> two contiguous 16B runs at byte offsets (kc*64+16g) and +32 within a row.
__device__ __forceinline__ v16h load_afrag(const _Float16* act, int strideHalves,
                                           int lrow, int g, int kc) {
  const char* base = reinterpret_cast<const char*>(act) +
                     (size_t)lrow * strideHalves * 2 + kc * 64 + 16 * g;
  v8h lo = *reinterpret_cast<const v8h*>(base);
  v8h hi = *reinterpret_cast<const v8h*>(base + 32);
  return __builtin_shufflevector(lo, hi, 0,1,2,3,4,5,6,7,8,9,10,11,12,13,14,15);
}

// B-fragment: pre-swizzled in workspace, 32 contiguous bytes per lane.
__device__ __forceinline__ v16h load_bfrag(const char* wgt, int fragByteOff, int lane) {
  return *reinterpret_cast<const v16h*>(wgt + fragByteOff + lane * 32);
}

__device__ __forceinline__ v8f wmma_f16(v16h a, v16h b, v8f c) {
  return __builtin_amdgcn_wmma_f32_16x16x32_f16(false, a, false, b, (short)0, c,
                                                false, false);
}

// One 128->128 layer for a wave's 16-row tile. in/out already offset to wave base.
__device__ __forceinline__ void mlp_layer128(const _Float16* in, _Float16* out,
                                             const char* wgt, int layerOff,
                                             const float* __restrict__ bias,
                                             int lane, bool leaky) {
  const int lrow = lane & 15, g = lane >> 4;
  v16h a0 = load_afrag(in, 128, lrow, g, 0);
  v16h a1 = load_afrag(in, 128, lrow, g, 1);
  v16h a2 = load_afrag(in, 128, lrow, g, 2);
  v16h a3 = load_afrag(in, 128, lrow, g, 3);
#pragma unroll
  for (int nt = 0; nt < 8; ++nt) {
    float bv = bias[nt * 16 + lrow];
    v8f acc = {bv, bv, bv, bv, bv, bv, bv, bv};
    acc = wmma_f16(a0, load_bfrag(wgt, layerOff + (0 * 8 + nt) * 1024, lane), acc);
    acc = wmma_f16(a1, load_bfrag(wgt, layerOff + (1 * 8 + nt) * 1024, lane), acc);
    acc = wmma_f16(a2, load_bfrag(wgt, layerOff + (2 * 8 + nt) * 1024, lane), acc);
    acc = wmma_f16(a3, load_bfrag(wgt, layerOff + (3 * 8 + nt) * 1024, lane), acc);
#pragma unroll
    for (int r = 0; r < 8; ++r) {
      float v = acc[r];
      if (leaky) v = (v > 0.f) ? v : 0.01f * v;
      // C/D layout: VGPR r, lane -> row r+8g, col nt*16+lrow
      out[(r + 8 * g) * 128 + nt * 16 + lrow] = (_Float16)v;
    }
  }
}

// ---------------------------------------------------------------------------
// Prep: w2c = inverse(c2w) (affine inverse), one thread per (b,v).
__global__ void prep_w2c(const float* __restrict__ src_cams, float* __restrict__ w2c) {
  int i = threadIdx.x;
  if (i >= B_ * NV) return;
  const float* m = src_cams + i * 27 + 11;  // c2w 4x4 row-major
  float a00 = m[0], a01 = m[1], a02 = m[2],  tx = m[3];
  float a10 = m[4], a11 = m[5], a12 = m[6],  ty = m[7];
  float a20 = m[8], a21 = m[9], a22 = m[10], tz = m[11];
  float c00 =  a11 * a22 - a12 * a21;
  float c01 = -(a10 * a22 - a12 * a20);
  float c02 =  a10 * a21 - a11 * a20;
  float id = 1.f / (a00 * c00 + a01 * c01 + a02 * c02);
  float i00 = c00 * id;
  float i01 = -(a01 * a22 - a02 * a21) * id;
  float i02 =  (a01 * a12 - a02 * a11) * id;
  float i10 = c01 * id;
  float i11 =  (a00 * a22 - a02 * a20) * id;
  float i12 = -(a00 * a12 - a02 * a10) * id;
  float i20 = c02 * id;
  float i21 = -(a00 * a21 - a01 * a20) * id;
  float i22 =  (a00 * a11 - a01 * a10) * id;
  float* o = w2c + i * 12;
  o[0] = i00; o[1] = i01; o[2]  = i02; o[3]  = -(i00 * tx + i01 * ty + i02 * tz);
  o[4] = i10; o[5] = i11; o[6]  = i12; o[7]  = -(i10 * tx + i11 * ty + i12 * tz);
  o[8] = i20; o[9] = i21; o[10] = i22; o[11] = -(i20 * tx + i21 * ty + i22 * tz);
}

// Prep: convert/pad weights to f16 and swizzle into B-fragment-major layout.
// Fragment (kc,nt): lane L -> n = nt*16 + L%16, g = L/16;
//                   half h -> k = kc*32 + (h<8 ? h : h+8) + 8*g
__global__ void prep_weights(const float* __restrict__ base_w,
                             const float* __restrict__ w1, const float* __restrict__ w2,
                             const float* __restrict__ w3, const float* __restrict__ w4,
                             const float* __restrict__ w5, char* __restrict__ wgt) {
  int tid = blockIdx.x * blockDim.x + threadIdx.x;
  const int total = 156 * 32;  // 24 + 4*32 + 4 fragments, 32 lanes each
  if (tid >= total) return;
  int lane = tid & 31;
  int frag = tid >> 5;
  const float* W;
  int off, kc, nt, NT, kReal, nReal, nStride;
  if (frag < 24)       { int f = frag;       W = base_w; off = OFF_L0; NT = 8; kc = f / 8; nt = f % 8; kReal = 70;  nReal = 128; nStride = 128; }
  else if (frag < 56)  { int f = frag - 24;  W = w1;     off = OFF_L1; NT = 8; kc = f / 8; nt = f % 8; kReal = 128; nReal = 128; nStride = 128; }
  else if (frag < 88)  { int f = frag - 56;  W = w2;     off = OFF_L2; NT = 8; kc = f / 8; nt = f % 8; kReal = 128; nReal = 128; nStride = 128; }
  else if (frag < 120) { int f = frag - 88;  W = w3;     off = OFF_L3; NT = 8; kc = f / 8; nt = f % 8; kReal = 128; nReal = 128; nStride = 128; }
  else if (frag < 152) { int f = frag - 120; W = w4;     off = OFF_L4; NT = 8; kc = f / 8; nt = f % 8; kReal = 128; nReal = 128; nStride = 128; }
  else                 { int f = frag - 152; W = w5;     off = OFF_L5; NT = 1; kc = f;     nt = 0;     kReal = 128; nReal = 3;   nStride = 3;   }
  _Float16* dst = (_Float16*)(wgt + off + (kc * NT + nt) * 1024 + lane * 32);
  int g = lane >> 4, nl = lane & 15;
  int n = nt * 16 + nl;
#pragma unroll
  for (int h = 0; h < 16; ++h) {
    int k = kc * 32 + (h < 8 ? h : h + 8) + 8 * g;
    float val = (k < kReal && n < nReal) ? W[(size_t)k * nStride + n] : 0.f;
    dst[h] = (_Float16)val;
  }
}

// ---------------------------------------------------------------------------
// Fused kernel: 256 threads (8 waves) handle 128 points.
// Phase 1: projection + bilinear sampling + masked mean/var -> LDS x (K=96 f16)
// Phase 2: base layer + 4 hidden layers + rgb head via WMMA; sigma via dot(pf).
__global__ __launch_bounds__(256) void nerf_fused(
    const float* __restrict__ xyz, const float* __restrict__ src_imgs,
    const float* __restrict__ src_cams, const float* __restrict__ src_feats,
    const float* __restrict__ base_b, const float* __restrict__ b1,
    const float* __restrict__ b2, const float* __restrict__ b3,
    const float* __restrict__ b4, const float* __restrict__ b5,
    const float* __restrict__ dens_w, const float* __restrict__ dens_b,
    const float* __restrict__ w2c_all, const char* __restrict__ wgt,
    float* __restrict__ out) {
  extern __shared__ char smem[];
  _Float16* s_pf = (_Float16*)smem;        // 128*128 f16 (pf, kept for sigma)
  _Float16* s_a  = s_pf + 128 * 128;       // ping
  _Float16* s_b  = s_a + 128 * 128;        // pong
  _Float16* s_x  = s_b;                    // x (128*96) aliases pong (barriered)

  const int t = threadIdx.x;

  // ---------------- phase 1: featurize 128 points, 2 threads/point ----------
  {
    const int p = t & 127;
    const int halfSel = t >> 7;            // 0: ch 0..17, 1: ch 18..34
    const int pt = blockIdx.x * 128 + p;
    const int b = pt / NPB;
    const float Px = xyz[(size_t)pt * 3 + 0];
    const float Py = xyz[(size_t)pt * 3 + 1];
    const float Pz = xyz[(size_t)pt * 3 + 2];
    const int c0 = halfSel ? 18 : 0;
    const int nc = halfSel ? 17 : 18;
    float S1[18], S2[18];
#pragma unroll
    for (int i = 0; i < 18; ++i) { S1[i] = 0.f; S2[i] = 0.f; }
    float M = 0.f;
    const float sx = (float)(FW - 1) / (float)(IMW - 1);
    const float sy = (float)(FH - 1) / (float)(IMH - 1);
    for (int v = 0; v < NV; ++v) {
      const float* wc = w2c_all + (b * NV + v) * 12;
      const float* cm = src_cams + (size_t)(b * NV + v) * 27;
      const float fx = cm[2], cxk = cm[4], fy = cm[6], cyk = cm[7];
      float X = wc[0] * Px + wc[1] * Py + wc[2]  * Pz + wc[3];
      float Y = wc[4] * Px + wc[5] * Py + wc[6]  * Pz + wc[7];
      float Z = wc[8] * Px + wc[9] * Py + wc[10] * Pz + wc[11];
      float depth = Z;
      float d = fmaxf(depth, 1e-8f);
      float px = (fx * X + cxk * Z) / d;
      float py = (fy * Y + cyk * Z) / d;
      bool inb = (depth > 1e-4f) && (px >= 0.f) && (px <= (float)(IMW - 1)) &&
                 (py >= 0.f) && (py <= (float)(IMH - 1));
      if (!inb) continue;
      M += 1.f;
      float x0f = floorf(px), y0f = floorf(py);
      float wx = px - x0f, wy = py - y0f;
      int ix0 = min(max((int)x0f, 0), IMW - 1);
      int ix1 = min(ix0 + 1, IMW - 1);
      int iy0 = min(max((int)y0f, 0), IMH - 1);
      int iy1 = min(iy0 + 1, IMH - 1);
      float w00 = (1.f - wx) * (1.f - wy), w01 = wx * (1.f - wy);
      float w10 = (1.f - wx) * wy, w11 = wx * wy;
      float pxf = px * sx, pyf = py * sy;
      float fx0f = floorf(pxf), fy0f = floorf(pyf);
      float fwx = pxf - fx0f, fwy = pyf - fy0f;
      int jx0 = min(max((int)fx0f, 0), FW - 1);
      int jx1 = min(jx0 + 1, FW - 1);
      int jy0 = min(max((int)fy0f, 0), FH - 1);
      int jy1 = min(jy0 + 1, FH - 1);
      float f00 = (1.f - fwx) * (1.f - fwy), f01 = fwx * (1.f - fwy);
      float f10 = (1.f - fwx) * fwy, f11 = fwx * fwy;
#pragma unroll
      for (int i = 0; i < 18; ++i) {
        if (i < nc) {
          int c = c0 + i;
          float f;
          if (c < 3) {
            const float* img =
                src_imgs + ((size_t)(b * NV + v) * 3 + c) * (size_t)(IMH * IMW);
            f = w00 * img[iy0 * IMW + ix0] + w01 * img[iy0 * IMW + ix1] +
                w10 * img[iy1 * IMW + ix0] + w11 * img[iy1 * IMW + ix1];
          } else {
            const float* ft = src_feats +
                ((size_t)(b * NV + v) * DFEAT + (c - 3)) * (size_t)(FH * FW);
            f = f00 * ft[jy0 * FW + jx0] + f01 * ft[jy0 * FW + jx1] +
                f10 * ft[jy1 * FW + jx0] + f11 * ft[jy1 * FW + jx1];
          }
          S1[i] += f;
          S2[i] += f * f;
        }
      }
    }
    const float inv = 1.f / (M + 1e-6f);
#pragma unroll
    for (int i = 0; i < 18; ++i) {
      if (i < nc) {
        int c = c0 + i;
        float mean = S1[i] * inv;
        float var = (S2[i] - 2.f * mean * S1[i] + mean * mean * M) * inv;
        s_x[p * 96 + c] = (_Float16)mean;
        s_x[p * 96 + 35 + c] = (_Float16)var;
      }
    }
    // zero pad K = 70..95 (13 entries per half-thread)
    for (int j = 70 + halfSel * 13; j < 83 + halfSel * 13; ++j)
      s_x[p * 96 + j] = (_Float16)0.f;
  }
  __syncthreads();

  // ---------------- phase 2: WMMA MLP, one 16-row tile per wave -------------
  const int lane = t & 31;
  const int wave = t >> 5;
  const int rowBase = wave * 16;
  const int lrow = lane & 15;
  const int g = lane >> 4;
  const int ptBase = blockIdx.x * 128 + rowBase;

  // base layer: x (K=96) -> pf (no activation; pf feeds both MLP and density)
  {
    const _Float16* in = s_x + rowBase * 96;
    _Float16* op = s_pf + rowBase * 128;
    v16h a0 = load_afrag(in, 96, lrow, g, 0);
    v16h a1 = load_afrag(in, 96, lrow, g, 1);
    v16h a2 = load_afrag(in, 96, lrow, g, 2);
#pragma unroll
    for (int nt = 0; nt < 8; ++nt) {
      float bv = base_b[nt * 16 + lrow];
      v8f acc = {bv, bv, bv, bv, bv, bv, bv, bv};
      acc = wmma_f16(a0, load_bfrag(wgt, OFF_L0 + (0 * 8 + nt) * 1024, lane), acc);
      acc = wmma_f16(a1, load_bfrag(wgt, OFF_L0 + (1 * 8 + nt) * 1024, lane), acc);
      acc = wmma_f16(a2, load_bfrag(wgt, OFF_L0 + (2 * 8 + nt) * 1024, lane), acc);
#pragma unroll
      for (int r = 0; r < 8; ++r)
        op[(r + 8 * g) * 128 + nt * 16 + lrow] = (_Float16)acc[r];
    }
  }
  __syncthreads();  // all s_x reads done before any wave writes s_b (aliased)

  mlp_layer128(s_pf + rowBase * 128, s_a + rowBase * 128, wgt, OFF_L1, b1, lane, true);
  mlp_layer128(s_a + rowBase * 128,  s_b + rowBase * 128, wgt, OFF_L2, b2, lane, true);
  mlp_layer128(s_b + rowBase * 128,  s_a + rowBase * 128, wgt, OFF_L3, b3, lane, true);
  mlp_layer128(s_a + rowBase * 128,  s_b + rowBase * 128, wgt, OFF_L4, b4, lane, true);

  // rgb head: K=128, single N tile (cols 0..2 real, 3..15 zero-padded)
  {
    const _Float16* in = s_b + rowBase * 128;
    v16h a0 = load_afrag(in, 128, lrow, g, 0);
    v16h a1 = load_afrag(in, 128, lrow, g, 1);
    v16h a2 = load_afrag(in, 128, lrow, g, 2);
    v16h a3 = load_afrag(in, 128, lrow, g, 3);
    float bv = (lrow < 3) ? b5[lrow] : 0.f;
    v8f acc = {bv, bv, bv, bv, bv, bv, bv, bv};
    acc = wmma_f16(a0, load_bfrag(wgt, OFF_L5 + 0 * 1024, lane), acc);
    acc = wmma_f16(a1, load_bfrag(wgt, OFF_L5 + 1 * 1024, lane), acc);
    acc = wmma_f16(a2, load_bfrag(wgt, OFF_L5 + 2 * 1024, lane), acc);
    acc = wmma_f16(a3, load_bfrag(wgt, OFF_L5 + 3 * 1024, lane), acc);
    if (lrow < 3) {
#pragma unroll
      for (int r = 0; r < 8; ++r) {
        int row = r + 8 * g;
        out[(size_t)(ptBase + row) * 4 + lrow] = tanhf(acc[r]);
      }
    }
  }

  // sigma = relu(pf . dens_w + dens_b), one row per low lane
  if (lane < 16) {
    const _Float16* pf = s_pf + (rowBase + lane) * 128;
    float acc = dens_b[0];
#pragma unroll 8
    for (int c = 0; c < 128; ++c) acc += (float)pf[c] * dens_w[c];
    out[(size_t)(ptBase + lane) * 4 + 3] = fmaxf(acc, 0.f);
  }
}

// ---------------------------------------------------------------------------
extern "C" void kernel_launch(void* const* d_in, const int* in_sizes, int n_in,
                              void* d_out, int out_size, void* d_ws, size_t ws_size,
                              hipStream_t stream) {
  const float* xyz       = (const float*)d_in[0];
  // d_in[1] = cam (unused by the reference computation)
  const float* src_imgs  = (const float*)d_in[2];
  const float* src_cams  = (const float*)d_in[3];
  const float* src_feats = (const float*)d_in[4];
  const float* base_w    = (const float*)d_in[5];
  const float* base_b    = (const float*)d_in[6];
  const float* w1 = (const float*)d_in[7];  const float* b1 = (const float*)d_in[8];
  const float* w2 = (const float*)d_in[9];  const float* b2 = (const float*)d_in[10];
  const float* w3 = (const float*)d_in[11]; const float* b3 = (const float*)d_in[12];
  const float* w4 = (const float*)d_in[13]; const float* b4 = (const float*)d_in[14];
  const float* w5 = (const float*)d_in[15]; const float* b5 = (const float*)d_in[16];
  const float* dens_w = (const float*)d_in[17];
  const float* dens_b = (const float*)d_in[18];
  float* outp = (float*)d_out;

  float* ws_w2c = (float*)d_ws;
  char* wgt = (char*)d_ws + WS_WGT_OFF;  // needs ~160KB of workspace

  prep_w2c<<<1, 64, 0, stream>>>(src_cams, ws_w2c);
  prep_weights<<<(156 * 32 + 255) / 256, 256, 0, stream>>>(base_w, w1, w2, w3, w4,
                                                           w5, wgt);
  const size_t smem = (size_t)3 * 128 * 128 * 2;  // 96KB: pf + ping + pong(x)
  nerf_fused<<<NPTS / 128, 256, smem, stream>>>(
      xyz, src_imgs, src_cams, src_feats, base_b, b1, b2, b3, b4, b5, dens_w,
      dens_b, ws_w2c, wgt, outp);
}
